// LocalAttention_3856880632104
// MI455X (gfx1250) — matrix operand order repaired and checked by reading, not compiled, not run on hardware
//
#include <hip/hip_runtime.h>
#include <hip/hip_bf16.h>
#include <math.h>

// Problem constants (match reference)
#define Bc  4
#define Sc  2048
#define Dc  512
#define Hc  8
#define Wc  16
#define DKc 64
#define BSc (Bc * Sc)   // 8192 rows

typedef __attribute__((ext_vector_type(16))) _Float16 v16h;
typedef __attribute__((ext_vector_type(8)))  _Float16 v8h;
typedef __attribute__((ext_vector_type(8)))  float    v8f;
typedef __attribute__((ext_vector_type(2)))  float    f32x2;
typedef __attribute__((ext_vector_type(2)))  _Float16 h2;
typedef __attribute__((ext_vector_type(4)))  int      i32x4;

// gfx1250 async global->LDS copy path (guarded; falls back to plain copies)
#if defined(__AMDGCN__) && __has_builtin(__builtin_amdgcn_global_load_async_to_lds_b128)
#define LA_ASYNC 1
typedef __attribute__((address_space(1))) i32x4 g1_i32x4;
typedef __attribute__((address_space(3))) i32x4 l3_i32x4;
#else
#define LA_ASYNC 0
#endif

__device__ __forceinline__ void la_wait_async() {
#if defined(__AMDGCN__)
#if __has_builtin(__builtin_amdgcn_s_wait_asynccnt)
  __builtin_amdgcn_s_wait_asynccnt(0);
#elif LA_ASYNC
  asm volatile("s_wait_asynccnt 0x0" ::: "memory");
#endif
#endif
}

// ---------------------------------------------------------------------------
// WMMA fragment builders (wave32, V_WMMA_F32_16X16X32_F16; A/B use this code's contiguous k order, not the ISA table order)
//   A (16x32 f16): lane -> row M = lane&15, K-half kh = lane>>4
//                  element e -> K = kh*16 + e
//   B (32x16 f16): lane -> col N = lane&15, K-half kh = lane>>4
//                  element e -> K = kh*16 + e
//   C (16x16 f32): lane -> col N = lane&15; vgpr r -> row M = 8*(lane>>4) + r
// ---------------------------------------------------------------------------

__device__ __forceinline__ v16h a_frag_f32(const float* __restrict__ X, int stride,
                                           int mb, int kb, int lane) {
  const int m  = lane & 15;
  const int kh = (lane >> 4) & 1;
  const float* base = X + (size_t)(mb + m) * stride + kb;
  v16h a;
#pragma unroll
  for (int j = 0; j < 8; ++j) {
    const int k = kh * 16 + j * 2;
    f32x2 x = *(const f32x2*)(base + k);
    a[2 * j]     = (_Float16)x[0];
    a[2 * j + 1] = (_Float16)x[1];
  }
  return a;
}

__device__ __forceinline__ v16h a_frag_f16(const _Float16* __restrict__ X, int stride,
                                           int mb, int kb, int lane) {
  const int m  = lane & 15;
  const int kh = (lane >> 4) & 1;
  const _Float16* base = X + (size_t)(mb + m) * stride + kb;
  v16h a;
#pragma unroll
  for (int j = 0; j < 8; ++j) {
    const int k = kh * 16 + j * 2;
    h2 x = *(const h2*)(base + k);
    a[2 * j]     = x[0];
    a[2 * j + 1] = x[1];
  }
  return a;
}

// B fragment from an N-major f16 source: src[n][stride] supplies B[k][n] = src[n0+n][kb+k]
__device__ __forceinline__ v16h b_frag_nmajor(const _Float16* __restrict__ src, int stride,
                                              int n0, int kb, int lane) {
  const int n  = lane & 15;
  const int kh = (lane >> 4) & 1;
  const _Float16* p = src + (size_t)(n0 + n) * stride + kb + kh * 16;
  v16h b;
#pragma unroll
  for (int j = 0; j < 8; ++j) {
    h2 x = *(const h2*)(p + 2 * j);
    b[2 * j]     = x[0];
    b[2 * j + 1] = x[1];
  }
  return b;
}

// B fragment from a K-major LDS tile src[48][64]: B[k][n] = src[key0+k][nb+n],
// row index clamped to the 48-row window (clamped rows carry softmax weight 0).
__device__ __forceinline__ v16h b_frag_kmajor_lds(const _Float16* src,
                                                  int key0, int nb, int lane) {
  const int n  = lane & 15;
  const int kh = (lane >> 4) & 1;
  v16h b;
#pragma unroll
  for (int e = 0; e < 16; ++e) {
    int k = key0 + kh * 16 + e;
    k = k > 47 ? 47 : k;
    b[e] = src[k * DKc + nb + n];
  }
  return b;
}

#define WMMA_F16(A, B, C) \
  __builtin_amdgcn_wmma_f32_16x16x32_f16(false, (A), false, (B), (short)0, (C), false, false)

// ---------------------------------------------------------------------------
// Kernel 1: cast + transpose a D x D f32 weight into f16 Wt[n][k] = W[k][n]
// ---------------------------------------------------------------------------
__global__ void la_wcast_kernel(const float* __restrict__ W, _Float16* __restrict__ Wt) {
  const int idx = blockIdx.x * blockDim.x + threadIdx.x;  // over D*D
  if (idx >= Dc * Dc) return;
  const int n = idx / Dc;
  const int k = idx - n * Dc;
  Wt[(size_t)n * Dc + k] = (_Float16)W[(size_t)k * Dc + n];
}

// ---------------------------------------------------------------------------
// Kernel 2: projection GEMM  Out[b,h,s,dk] = f16( X[m,:] @ W + bias ), X f32.
// 1 wave per 16(m) x 64(n) tile; K = 512 in 16 chunks of 32, double-buffered.
// ---------------------------------------------------------------------------
__global__ void __launch_bounds__(32)
la_proj_kernel(const float* __restrict__ X, const _Float16* __restrict__ Wt,
               const float* __restrict__ bias, _Float16* __restrict__ Out) {
  const int lane = threadIdx.x;
  const int mb = blockIdx.x * 16;   // [0, B*S)
  const int nb = blockIdx.y * 64;   // [0, D)
  const int nlane = lane & 15;

  v8f acc[4];
#pragma unroll
  for (int t = 0; t < 4; ++t) {
    const float bv = bias[nb + t * 16 + nlane];
#pragma unroll
    for (int r = 0; r < 8; ++r) acc[t][r] = bv;
  }

  // software-pipelined k-loop: fetch chunk i+1 while multiplying chunk i
  v16h a_cur = a_frag_f32(X, Dc, mb, 0, lane);
  v16h b_cur[4];
#pragma unroll
  for (int t = 0; t < 4; ++t) b_cur[t] = b_frag_nmajor(Wt, Dc, nb + t * 16, 0, lane);

  const int NK = Dc / 32;  // 16
#pragma unroll
  for (int i = 0; i < NK; ++i) {
    v16h a_nxt = a_cur;
    v16h b_nxt[4] = {b_cur[0], b_cur[1], b_cur[2], b_cur[3]};
    if (i + 1 < NK) {
      const int kn = (i + 1) * 32;
      a_nxt = a_frag_f32(X, Dc, mb, kn, lane);
#pragma unroll
      for (int t = 0; t < 4; ++t) b_nxt[t] = b_frag_nmajor(Wt, Dc, nb + t * 16, kn, lane);
    }
#pragma unroll
    for (int t = 0; t < 4; ++t) acc[t] = WMMA_F16(a_cur, b_cur[t], acc[t]);
    a_cur = a_nxt;
#pragma unroll
    for (int t = 0; t < 4; ++t) b_cur[t] = b_nxt[t];
  }

  const int mrow = mb + ((lane >> 4) << 3);
#pragma unroll
  for (int t = 0; t < 4; ++t) {
    const int n  = nb + t * 16 + nlane;
    const int h  = n >> 6;       // n / DK
    const int dk = n & 63;
#pragma unroll
    for (int r = 0; r < 8; ++r) {
      const int m = mrow + r;
      const int bi = m >> 11;    // m / S
      const int s  = m & (Sc - 1);
      Out[(((size_t)bi * Hc + h) * Sc + s) * DKc + dk] = (_Float16)acc[t][r];
    }
  }
}

// ---------------------------------------------------------------------------
// Kernel 3: banded attention.  1 wave per (b,h, 16-query tile).
// Keys live in a 48-wide window [qs-16, qs+31]; band mask |q-k| <= 16.
// K/V windows staged to LDS (async global->LDS when available).
// ---------------------------------------------------------------------------
__global__ void __launch_bounds__(32)
la_attn_kernel(const _Float16* __restrict__ Q, const _Float16* __restrict__ K,
               const _Float16* __restrict__ V, _Float16* __restrict__ ctx) {
  __shared__ _Float16 Kw[48][DKc];     // 6 KB
  __shared__ _Float16 Vw[48][DKc];     // 6 KB
  __shared__ float    sc_lds[16][48];  // 3 KB
  __shared__ _Float16 p_lds[16][64];   // 2 KB, cols 48..63 zero-padded

  const int lane = threadIdx.x;
  const int nqt  = Sc / 16;                 // 128
  const int qt   = blockIdx.x % nqt;
  const int bh   = blockIdx.x / nqt;        // b*H + h
  const int h    = bh & (Hc - 1);
  const int b    = bh >> 3;
  const int qs   = qt * 16;
  const int k0   = qs - Wc;                 // window start (may be negative)

  const _Float16* Qb = Q + (size_t)bh * Sc * DKc;
  const _Float16* Kb = K + (size_t)bh * Sc * DKc;
  const _Float16* Vb = V + (size_t)bh * Sc * DKc;

  // ---- stage K/V 48x64 windows into LDS (16B per lane per step) ----
#pragma unroll
  for (int c = lane; c < 384; c += 32) {   // 48 rows * 8 chunks/row, 12 iters
    const int row = c >> 3;                // 8 x 16B chunks per 128B row
    const int col = (c & 7) * 8;           // in halves
    int key = k0 + row;
    key = key < 0 ? 0 : (key >= Sc ? Sc - 1 : key);
    const _Float16* gK = Kb + (size_t)key * DKc + col;
    const _Float16* gV = Vb + (size_t)key * DKc + col;
#if LA_ASYNC
    __builtin_amdgcn_global_load_async_to_lds_b128(
        (g1_i32x4*)gK, (l3_i32x4*)&Kw[row][col], 0, 0);
    __builtin_amdgcn_global_load_async_to_lds_b128(
        (g1_i32x4*)gV, (l3_i32x4*)&Vw[row][col], 0, 0);
#else
    *(v8h*)&Kw[row][col] = *(const v8h*)gK;
    *(v8h*)&Vw[row][col] = *(const v8h*)gV;
#endif
  }
#if LA_ASYNC
  la_wait_async();
#endif
  __syncthreads();

  // ---- scores: S[16q x 48k] = Q(16x64) @ K^T ----
  v16h aq[2];
  aq[0] = a_frag_f16(Qb, DKc, qs, 0, lane);
  aq[1] = a_frag_f16(Qb, DKc, qs, 32, lane);

  v8f sc[3];
#pragma unroll
  for (int t = 0; t < 3; ++t)
#pragma unroll
    for (int r = 0; r < 8; ++r) sc[t][r] = 0.0f;

#pragma unroll
  for (int t = 0; t < 3; ++t) {
#pragma unroll
    for (int c = 0; c < 2; ++c) {
      v16h bk = b_frag_nmajor(&Kw[0][0], DKc, t * 16, c * 32, lane);
      sc[t] = WMMA_F16(aq[c], bk, sc[t]);
    }
  }

  // ---- scale + band mask -> LDS ----
  const float scale = 0.125f;  // 1/sqrt(DK)
  const int qrb   = (lane >> 4) << 3;
  const int nlane = lane & 15;
#pragma unroll
  for (int t = 0; t < 3; ++t) {
#pragma unroll
    for (int r = 0; r < 8; ++r) {
      const int q   = qs + qrb + r;
      const int key = k0 + t * 16 + nlane;
      const bool ok = (key >= 0) & (key < Sc) & (key >= q - Wc) & (key <= q + Wc);
      sc_lds[qrb + r][t * 16 + nlane] = ok ? sc[t][r] * scale : -3.4e38f;
    }
  }
  __syncthreads();

  // ---- row softmax (lanes 0..15, one query row each) ----
  if (lane < 16) {
    float mx = -3.4e38f;
#pragma unroll 4
    for (int c = 0; c < 48; ++c) mx = fmaxf(mx, sc_lds[lane][c]);
    float sum = 0.0f;
#pragma unroll 4
    for (int c = 0; c < 48; ++c) sum += __expf(sc_lds[lane][c] - mx);
    const float inv = 1.0f / sum;
#pragma unroll 4
    for (int c = 0; c < 48; ++c)
      p_lds[lane][c] = (_Float16)(__expf(sc_lds[lane][c] - mx) * inv);
#pragma unroll
    for (int c = 48; c < 64; ++c) p_lds[lane][c] = (_Float16)0.0f;
  }
  __syncthreads();

  // ---- ctx(16x64) = P(16x64, zero-padded) @ V(window x 64) ----
  v16h ap[2];
  ap[0] = a_frag_f16(&p_lds[0][0], 64, 0, 0, lane);
  ap[1] = a_frag_f16(&p_lds[0][0], 64, 0, 32, lane);

  v8f ct[4];
#pragma unroll
  for (int t = 0; t < 4; ++t)
#pragma unroll
    for (int r = 0; r < 8; ++r) ct[t][r] = 0.0f;

#pragma unroll
  for (int t = 0; t < 4; ++t) {
#pragma unroll
    for (int c = 0; c < 2; ++c) {
      v16h bv = b_frag_kmajor_lds(&Vw[0][0], c * 32, t * 16, lane);
      ct[t] = WMMA_F16(ap[c], bv, ct[t]);
    }
  }

  // ---- store ctx as f16 in [b, s, d] ----
#pragma unroll
  for (int t = 0; t < 4; ++t) {
    const int d = h * DKc + t * 16 + nlane;
#pragma unroll
    for (int r = 0; r < 8; ++r) {
      const int q = qs + qrb + r;
      ctx[((size_t)b * Sc + q) * Dc + d] = (_Float16)ct[t][r];
    }
  }
}

// ---------------------------------------------------------------------------
// Kernel 4: output projection  out[m, n] = ctx_f16[m,:] @ Wo + bo  (f32 out)
// ---------------------------------------------------------------------------
__global__ void __launch_bounds__(32)
la_oproj_kernel(const _Float16* __restrict__ X, const _Float16* __restrict__ Wt,
                const float* __restrict__ bias, float* __restrict__ Out) {
  const int lane = threadIdx.x;
  const int mb = blockIdx.x * 16;
  const int nb = blockIdx.y * 64;
  const int nlane = lane & 15;

  v8f acc[4];
#pragma unroll
  for (int t = 0; t < 4; ++t) {
    const float bv = bias[nb + t * 16 + nlane];
#pragma unroll
    for (int r = 0; r < 8; ++r) acc[t][r] = bv;
  }

  v16h a_cur = a_frag_f16(X, Dc, mb, 0, lane);
  v16h b_cur[4];
#pragma unroll
  for (int t = 0; t < 4; ++t) b_cur[t] = b_frag_nmajor(Wt, Dc, nb + t * 16, 0, lane);

  const int NK = Dc / 32;  // 16
#pragma unroll
  for (int i = 0; i < NK; ++i) {
    v16h a_nxt = a_cur;
    v16h b_nxt[4] = {b_cur[0], b_cur[1], b_cur[2], b_cur[3]};
    if (i + 1 < NK) {
      const int kn = (i + 1) * 32;
      a_nxt = a_frag_f16(X, Dc, mb, kn, lane);
#pragma unroll
      for (int t = 0; t < 4; ++t) b_nxt[t] = b_frag_nmajor(Wt, Dc, nb + t * 16, kn, lane);
    }
#pragma unroll
    for (int t = 0; t < 4; ++t) acc[t] = WMMA_F16(a_cur, b_cur[t], acc[t]);
    a_cur = a_nxt;
#pragma unroll
    for (int t = 0; t < 4; ++t) b_cur[t] = b_nxt[t];
  }

  const int mrow = mb + ((lane >> 4) << 3);
#pragma unroll
  for (int t = 0; t < 4; ++t) {
    const int n = nb + t * 16 + nlane;
#pragma unroll
    for (int r = 0; r < 8; ++r) {
      const int m = mrow + r;
      Out[(size_t)m * Dc + n] = acc[t][r];
    }
  }
}

// ---------------------------------------------------------------------------
// Launch: weights->f16^T, 3 projections, banded attention, output projection.
// Workspace: 4 transposed f16 weights (1M f16) + Q/K/V/ctx f16 (16M f16) = 34 MB.
// ---------------------------------------------------------------------------
extern "C" void kernel_launch(void* const* d_in, const int* in_sizes, int n_in,
                              void* d_out, int out_size, void* d_ws, size_t ws_size,
                              hipStream_t stream) {
  const float* query = (const float*)d_in[0];
  const float* key   = (const float*)d_in[1];
  const float* value = (const float*)d_in[2];
  const float* Wq    = (const float*)d_in[3];
  const float* bq    = (const float*)d_in[4];
  const float* Wk    = (const float*)d_in[5];
  const float* bk    = (const float*)d_in[6];
  const float* Wv    = (const float*)d_in[7];
  const float* bv    = (const float*)d_in[8];
  const float* Wo    = (const float*)d_in[9];
  const float* bo    = (const float*)d_in[10];
  float* out = (float*)d_out;

  _Float16* ws = (_Float16*)d_ws;
  const size_t WSZ = (size_t)Dc * Dc;             // 262144
  const size_t TSZ = (size_t)Bc * Hc * Sc * DKc;  // 4194304
  _Float16* wq_t = ws;
  _Float16* wk_t = wq_t + WSZ;
  _Float16* wv_t = wk_t + WSZ;
  _Float16* wo_t = wv_t + WSZ;
  _Float16* Qh   = wo_t + WSZ;
  _Float16* Kh   = Qh + TSZ;
  _Float16* Vh   = Kh + TSZ;
  _Float16* Ch   = Vh + TSZ;

  // 1) cast + transpose weights to f16 [n][k]
  {
    const int threads = 256;
    const int blocks  = (Dc * Dc + threads - 1) / threads;
    la_wcast_kernel<<<blocks, threads, 0, stream>>>(Wq, wq_t);
    la_wcast_kernel<<<blocks, threads, 0, stream>>>(Wk, wk_t);
    la_wcast_kernel<<<blocks, threads, 0, stream>>>(Wv, wv_t);
    la_wcast_kernel<<<blocks, threads, 0, stream>>>(Wo, wo_t);
  }

  // 2) QKV projections (WMMA f16, f32 accumulate) -> [B,H,S,DK] f16
  dim3 pg(BSc / 16, Dc / 64);
  la_proj_kernel<<<pg, 32, 0, stream>>>(query, wq_t, bq, Qh);
  la_proj_kernel<<<pg, 32, 0, stream>>>(key,   wk_t, bk, Kh);
  la_proj_kernel<<<pg, 32, 0, stream>>>(value, wv_t, bv, Vh);

  // 3) banded attention (48-key window per 16-query tile)
  la_attn_kernel<<<Bc * Hc * (Sc / 16), 32, 0, stream>>>(Qh, Kh, Vh, Ch);

  // 4) output projection -> f32 d_out
  la_oproj_kernel<<<pg, 32, 0, stream>>>(Ch, wo_t, bo, out);
}
